// cosFormerBlock_9783935500304
// MI455X (gfx1250) — compile-verified, matrix operand order checked by reading, not run on hardware
//
#include <hip/hip_runtime.h>

// ---------------------------------------------------------------------------
// cosFormer block for MI455X (gfx1250)
//  - all GEMMs on v_wmma_f32_16x16x32_bf16 (f32 accumulate)
//  - B operands pre-transposed so A and B tiles stage identically
//  - double-buffered LDS staging via GLOBAL_LOAD_ASYNC_TO_LDS_B128
//    (ASYNCcnt-tracked), sync fallback if the builtin is absent
// ---------------------------------------------------------------------------

typedef __bf16 bf16;
typedef __attribute__((ext_vector_type(16))) __bf16 v16bf;
typedef __attribute__((ext_vector_type(8)))  __bf16 v8bf;
typedef __attribute__((ext_vector_type(8)))  float  v8f;

static constexpr int Bc  = 4;
static constexpr int Tc  = 4096;
static constexpr int Ec  = 1024;
static constexpr int Hc  = 16;
static constexpr int Dc  = 64;
static constexpr int BTc = Bc * Tc;     // 16384 rows
static constexpr int E4c = 4 * Ec;      // 4096
static constexpr int BHc = Bc * Hc;     // 64
static constexpr int D2c = 2 * Dc;      // 128

__device__ __forceinline__ bf16 f2bf(float f) {
  unsigned u = __builtin_bit_cast(unsigned, f);
  unsigned r = u + 0x7FFFu + ((u >> 16) & 1u);  // round to nearest even
  unsigned short h = (unsigned short)(r >> 16);
  return __builtin_bit_cast(bf16, h);
}
__device__ __forceinline__ float bf2f(bf16 b) {
  unsigned u = ((unsigned)__builtin_bit_cast(unsigned short, b)) << 16;
  return __builtin_bit_cast(float, u);
}

// ---- CDNA5 async memory->LDS path ------------------------------------------
#if defined(__gfx1250__) && __has_builtin(__builtin_amdgcn_global_load_async_to_lds_b128)
#define HAVE_ASYNC_LDS 1
#else
#define HAVE_ASYNC_LDS 0
#endif

#if HAVE_ASYNC_LDS
// builtin signature (from hipcc diagnostic): param 1 is
//   '__attribute__((vector_size(16))) int __device__ *'   (AS1 = global)
// param 2 is the LDS destination (AS3), then imm offset, imm cpol.
typedef int v4i_vs __attribute__((vector_size(16)));
typedef __attribute__((address_space(1))) v4i_vs* gptr128;
typedef __attribute__((address_space(3))) v4i_vs* lptr128;
#if __has_builtin(__builtin_amdgcn_s_wait_asynccnt)
#define ASYNC_WAIT(n) __builtin_amdgcn_s_wait_asynccnt(n)
#else
#define ASYNC_WAIT(n) asm volatile("s_wait_asynccnt %0" ::"n"(n) : "memory")
#endif
#else
#define ASYNC_WAIT(n) ((void)0)
#endif

__device__ __forceinline__ void copy16_to_lds(const bf16* g, bf16* l) {
#if HAVE_ASYNC_LDS
  __builtin_amdgcn_global_load_async_to_lds_b128((gptr128)(g), (lptr128)(l), 0,
                                                 0);
#else
  *(uint4*)l = *(const uint4*)g;
#endif
}

// ---------------------------------------------------------------------------
// fp32 [K x N] -> bf16 transposed [N x K] (weights)
// ---------------------------------------------------------------------------
__global__ __launch_bounds__(256) void cvtT_kernel(const float* __restrict__ in,
                                                   bf16* __restrict__ out,
                                                   int K, int N) {
  __shared__ float tile[32][33];
  const int kb = blockIdx.y * 32, nb = blockIdx.x * 32;
  const int tx = threadIdx.x & 31, ty = threadIdx.x >> 5;  // 8 rows / pass
#pragma unroll
  for (int i = 0; i < 32; i += 8)
    tile[ty + i][tx] = in[(size_t)(kb + ty + i) * N + nb + tx];
  __syncthreads();
#pragma unroll
  for (int i = 0; i < 32; i += 8)
    out[(size_t)(nb + ty + i) * K + kb + tx] = f2bf(tile[tx][ty + i]);
}

// ---------------------------------------------------------------------------
// v[B,T,E] bf16 -> vt[BH, D, T] bf16 (per-head transpose for K'^T V GEMM)
// ---------------------------------------------------------------------------
__global__ __launch_bounds__(256) void vtr_kernel(const bf16* __restrict__ v,
                                                  bf16* __restrict__ vt) {
  __shared__ bf16 tile[32][33];
  const int bh = blockIdx.z, b = bh >> 4, h = bh & 15;
  const int t0 = blockIdx.x * 32, d0 = blockIdx.y * 32;
  const int tx = threadIdx.x & 31, ty = threadIdx.x >> 5;
  const bf16* src = v + (size_t)b * Tc * Ec + (size_t)h * Dc;
#pragma unroll
  for (int i = 0; i < 32; i += 8)
    tile[ty + i][tx] = src[(size_t)(t0 + ty + i) * Ec + d0 + tx];
  __syncthreads();
  bf16* dst = vt + (size_t)bh * Dc * Tc;
#pragma unroll
  for (int i = 0; i < 32; i += 8)
    dst[(size_t)(d0 + ty + i) * Tc + t0 + tx] = tile[tx][ty + i];
}

// ---------------------------------------------------------------------------
// LayerNorm over E=1024, one row per block, bf16 out
// ---------------------------------------------------------------------------
__global__ __launch_bounds__(256) void ln_kernel(const float* __restrict__ x,
                                                 const float* __restrict__ w,
                                                 const float* __restrict__ b,
                                                 bf16* __restrict__ out) {
  __shared__ float red[256];
  const int tid = threadIdx.x;
  const size_t row = blockIdx.x;
  const float* xr = x + row * Ec;

  float4 xv = *(const float4*)(xr + tid * 4);
  red[tid] = xv.x + xv.y + xv.z + xv.w;
  __syncthreads();
  for (int o = 128; o > 0; o >>= 1) {
    if (tid < o) red[tid] += red[tid + o];
    __syncthreads();
  }
  const float mu = red[0] * (1.0f / Ec);
  __syncthreads();

  float d0 = xv.x - mu, d1 = xv.y - mu, d2 = xv.z - mu, d3 = xv.w - mu;
  red[tid] = d0 * d0 + d1 * d1 + d2 * d2 + d3 * d3;
  __syncthreads();
  for (int o = 128; o > 0; o >>= 1) {
    if (tid < o) red[tid] += red[tid + o];
    __syncthreads();
  }
  const float rs = rsqrtf(red[0] * (1.0f / Ec) + 1e-5f);

  const int e = tid * 4;
  bf16* orow = out + row * Ec;
  orow[e + 0] = f2bf(d0 * rs * w[e + 0] + b[e + 0]);
  orow[e + 1] = f2bf(d1 * rs * w[e + 1] + b[e + 1]);
  orow[e + 2] = f2bf(d2 * rs * w[e + 2] + b[e + 2]);
  orow[e + 3] = f2bf(d3 * rs * w[e + 3] + b[e + 3]);
}

// ---------------------------------------------------------------------------
// WMMA GEMM:  C = epi(A @ B + bias) [+resid] [*rowscale]
//   A:  [M x K] bf16 row-major (lda)
//   Bt: [N x K] bf16 row-major (ldb)  == B transposed
//   Tile 128x128x32, 8 waves (2x4), wave tile 64x32 -> 4x2 WMMA 16x16x32.
//   Double-buffered, async staging, software-pipelined with peeled tail.
//   Batched: z -> (zb=z/zdivH, zh=z%zdivH), offsets s*1*zb + s*2*zh.
//   Cb optionally stored transposed (cbT).  ACT: 0 none, 1 relu, 2 exact gelu
// ---------------------------------------------------------------------------
static constexpr int BM = 128, BN = 128, BK = 32;
static constexpr int LDSP = 40;  // bf16 pitch (80B rows, 16B aligned)

template <int ACT>
__global__ __launch_bounds__(256) void gemm_kernel(
    const bf16* __restrict__ A, const bf16* __restrict__ Bt,
    const float* __restrict__ bias, const float* __restrict__ resid,
    const float* __restrict__ rowscale, float* __restrict__ Cf,
    bf16* __restrict__ Cb, int M, int N, int K, int lda, int ldb, int ldc,
    int ldcb, int cbT, int zdivH, long long sA1, long long sA2, long long sB1,
    long long sB2, long long sC1, long long sC2, long long sCb1,
    long long sCb2, long long sS1, long long sS2) {
  __shared__ __attribute__((aligned(16))) bf16 ldsA[2][BM * LDSP];
  __shared__ __attribute__((aligned(16))) bf16 ldsB[2][BN * LDSP];

  const int z = blockIdx.z;
  const int zb = z / zdivH, zh = z % zdivH;
  A += zb * sA1 + zh * sA2;
  Bt += zb * sB1 + zh * sB2;
  const long long offC = zb * sC1 + zh * sC2;
  const long long offCb = zb * sCb1 + zh * sCb2;
  const long long offS = zb * sS1 + zh * sS2;

  const int n0 = blockIdx.x * BN;
  const int m0 = blockIdx.y * BM;
  const int tid = threadIdx.x;
  const int lane = tid & 31;
  const int wave = tid >> 5;
  const int wm = wave >> 2;  // 0..1 -> 64 rows
  const int wn = wave & 3;   // 0..3 -> 32 cols
  const int kb = (lane >= 16) ? 8 : 0;
  const int l15 = lane & 15;

  // per-thread staging chunks: 2 x 16B per tile per thread, rows clamped so
  // every copy is in-bounds (junk rows only feed discarded outputs; K%32==0)
  const bf16* ap[2];
  const bf16* bp[2];
  bf16* la[2][2];
  bf16* lb[2][2];
#pragma unroll
  for (int p = 0; p < 2; ++p) {
    int idx = (p * 256 + tid) * 8;
    int srow = idx >> 5, scol = idx & 31;
    int gm = m0 + srow;
    int gn = n0 + srow;
    ap[p] = A + (size_t)((gm < M) ? gm : (M - 1)) * lda + scol;
    bp[p] = Bt + (size_t)((gn < N) ? gn : (N - 1)) * ldb + scol;
#pragma unroll
    for (int s = 0; s < 2; ++s) {
      la[s][p] = &ldsA[s][srow * LDSP + scol];
      lb[s][p] = &ldsB[s][srow * LDSP + scol];
    }
  }

  v8f acc[4][2] = {};

  // stage one 128x32 A tile + 128x32 Bt tile into LDS stage s, advance ptrs
  auto stage = [&](int s) {
#pragma unroll
    for (int p = 0; p < 2; ++p) {
      copy16_to_lds(ap[p], la[s][p]);
      ap[p] += BK;
    }
#pragma unroll
    for (int p = 0; p < 2; ++p) {
      copy16_to_lds(bp[p], lb[s][p]);
      bp[p] += BK;
    }
  };

  // fragment loads + 8 WMMAs from LDS stage s
  auto compute = [&](int s) {
    v16bf afr[4], bfr[2];
#pragma unroll
    for (int i = 0; i < 4; ++i) {
      const bf16* base = &ldsA[s][(wm * 64 + i * 16 + l15) * LDSP + kb];
      v8bf lo = *(const v8bf*)(base);
      v8bf hi = *(const v8bf*)(base + 16);
      afr[i] = __builtin_shufflevector(lo, hi, 0, 1, 2, 3, 4, 5, 6, 7, 8, 9,
                                       10, 11, 12, 13, 14, 15);
    }
#pragma unroll
    for (int j = 0; j < 2; ++j) {
      const bf16* base = &ldsB[s][(wn * 32 + j * 16 + l15) * LDSP + kb];
      v8bf lo = *(const v8bf*)(base);
      v8bf hi = *(const v8bf*)(base + 16);
      bfr[j] = __builtin_shufflevector(lo, hi, 0, 1, 2, 3, 4, 5, 6, 7, 8, 9,
                                       10, 11, 12, 13, 14, 15);
    }
#pragma unroll
    for (int i = 0; i < 4; ++i)
#pragma unroll
      for (int j = 0; j < 2; ++j)
        acc[i][j] = __builtin_amdgcn_wmma_f32_16x16x32_bf16(
            false, afr[i], false, bfr[j], (short)0, acc[i][j], false, false);
  };

  // software pipeline: prologue stage, steady-state, peeled tail
  stage(0);
  int buf = 0;
  for (int k0 = 0; k0 < K - BK; k0 += BK) {
    stage(buf ^ 1);
    ASYNC_WAIT(4);  // previous stage's 4 copies complete (in-order counter)
    __builtin_prefetch(ap[0], 0, 1);  // warm L2 one stage further ahead
    __syncthreads();
    compute(buf);
    __syncthreads();
    buf ^= 1;
  }
  ASYNC_WAIT(0);
  __syncthreads();
  compute(buf);

  // ---- epilogue ----
  const int hi8 = (lane >= 16) ? 8 : 0;
#pragma unroll
  for (int i = 0; i < 4; ++i) {
#pragma unroll
    for (int j = 0; j < 2; ++j) {
#pragma unroll
      for (int r = 0; r < 8; ++r) {
        int gm = m0 + wm * 64 + i * 16 + r + hi8;
        int gn = n0 + wn * 32 + j * 16 + l15;
        if (gm < M && gn < N) {
          float v = acc[i][j][r];
          if (bias) v += bias[gn];
          if (ACT == 1) v = fmaxf(v, 0.0f);
          if (ACT == 2) v = 0.5f * v * (1.0f + erff(v * 0.70710678118654752f));
          long long ci = offC + (long long)gm * ldc + gn;
          if (resid) v += resid[ci];
          if (rowscale) v *= rowscale[offS + gm];
          if (Cf) Cf[ci] = v;
          if (Cb) {
            long long cbi = offCb + (cbT ? ((long long)gn * ldcb + gm)
                                         : ((long long)gm * ldcb + gn));
            Cb[cbi] = f2bf(v);
          }
        }
      }
    }
  }
}

// ---------------------------------------------------------------------------
// cosine reweighting: q,k [B,T,E] bf16 (post-relu) ->
//   qr[BH,T,2D] = [q*sin, q*cos],  kr[BH,2D,T] (transposed)
// ---------------------------------------------------------------------------
__global__ __launch_bounds__(256) void reweight_kernel(
    const bf16* __restrict__ q, const bf16* __restrict__ k,
    bf16* __restrict__ qr, bf16* __restrict__ kr) {
  const int bt = blockIdx.x;
  const int b = bt >> 12;  // / Tc
  const int t = bt & (Tc - 1);
  const float idx = (1.5707963267948966f / Tc) * (float)(t + 1);
  const float s = sinf(idx), c = cosf(idx);
#pragma unroll
  for (int i = 0; i < 4; ++i) {
    int e = threadIdx.x + 256 * i;
    int h = e >> 6, d = e & 63;
    size_t src = (size_t)bt * Ec + e;
    float qv = bf2f(q[src]);
    float kv = bf2f(k[src]);
    size_t bh = (size_t)b * Hc + h;
    size_t qbase = (bh * Tc + t) * D2c;
    qr[qbase + d] = f2bf(qv * s);
    qr[qbase + Dc + d] = f2bf(qv * c);
    size_t kbase = bh * (size_t)D2c * Tc;
    kr[kbase + (size_t)d * Tc + t] = f2bf(kv * s);
    kr[kbase + (size_t)(Dc + d) * Tc + t] = f2bf(kv * c);
  }
}

// ---------------------------------------------------------------------------
// ksum[bh][m] = sum_t kr[bh][m][t]
// ---------------------------------------------------------------------------
__global__ __launch_bounds__(128) void colsum_kernel(const bf16* __restrict__ kr,
                                                     float* __restrict__ ksum) {
  const int bh = blockIdx.x;
  const int m = threadIdx.x;  // 0..127
  const bf16* row = kr + ((size_t)bh * D2c + m) * Tc;
  float s = 0.0f;
  for (int t = 0; t < Tc; t += 8) {
    v8bf v = *(const v8bf*)(row + t);
#pragma unroll
    for (int j = 0; j < 8; ++j) s += bf2f(v[j]);
  }
  ksum[(size_t)bh * D2c + m] = s;
}

// ---------------------------------------------------------------------------
// zr[bh][t] = 1 / max(q'[bh,t,:] . ksum[bh,:], 1e-6)
// ---------------------------------------------------------------------------
__global__ __launch_bounds__(256) void zden_kernel(const bf16* __restrict__ qr,
                                                   const float* __restrict__ ksum,
                                                   float* __restrict__ zr) {
  __shared__ float ks[D2c];
  const int bh = blockIdx.y;
  if (threadIdx.x < D2c) ks[threadIdx.x] = ksum[(size_t)bh * D2c + threadIdx.x];
  __syncthreads();
  const int t = blockIdx.x * 256 + threadIdx.x;
  const bf16* row = qr + ((size_t)bh * Tc + t) * D2c;
  float s = 0.0f;
#pragma unroll 16
  for (int m = 0; m < D2c; ++m) s += bf2f(row[m]) * ks[m];
  zr[(size_t)bh * Tc + t] = 1.0f / fmaxf(s, 1e-6f);
}

// ---------------------------------------------------------------------------
// Host-side orchestration
// ---------------------------------------------------------------------------
extern "C" void kernel_launch(void* const* d_in, const int* in_sizes, int n_in,
                              void* d_out, int out_size, void* d_ws,
                              size_t ws_size, hipStream_t stream) {
  (void)in_sizes; (void)n_in; (void)out_size; (void)ws_size;

  const float* x    = (const float*)d_in[0];
  const float* ln1w = (const float*)d_in[1];
  const float* ln1b = (const float*)d_in[2];
  const float* wq   = (const float*)d_in[3];
  const float* bq   = (const float*)d_in[4];
  const float* wk   = (const float*)d_in[5];
  const float* bk   = (const float*)d_in[6];
  const float* wv   = (const float*)d_in[7];
  const float* bv   = (const float*)d_in[8];
  const float* wo   = (const float*)d_in[9];
  const float* bo   = (const float*)d_in[10];
  const float* ln2w = (const float*)d_in[11];
  const float* ln2b = (const float*)d_in[12];
  const float* wfc  = (const float*)d_in[13];
  const float* bfc  = (const float*)d_in[14];
  const float* wpr  = (const float*)d_in[15];
  const float* bpr  = (const float*)d_in[16];
  float* out = (float*)d_out;

  char* ws = (char*)d_ws;
  size_t off = 0;
  auto alloc = [&](size_t bytes) -> char* {
    char* p = ws + off;
    off += (bytes + 255) & ~(size_t)255;
    return p;
  };
  const size_t EE = (size_t)Ec * Ec;
  bf16* wq_t  = (bf16*)alloc(EE * 2);
  bf16* wk_t  = (bf16*)alloc(EE * 2);
  bf16* wv_t  = (bf16*)alloc(EE * 2);
  bf16* wo_t  = (bf16*)alloc(EE * 2);
  bf16* wfc_t = (bf16*)alloc((size_t)Ec * E4c * 2);
  bf16* wpr_t = (bf16*)alloc((size_t)E4c * Ec * 2);
  bf16* h1_b  = (bf16*)alloc((size_t)BTc * Ec * 2);
  bf16* q_b   = (bf16*)alloc((size_t)BTc * Ec * 2);
  bf16* k_b   = (bf16*)alloc((size_t)BTc * Ec * 2);
  bf16* v_b   = (bf16*)alloc((size_t)BTc * Ec * 2);
  bf16* vt_b  = (bf16*)alloc((size_t)BHc * Dc * Tc * 2);
  bf16* qr_b  = (bf16*)alloc((size_t)BHc * Tc * D2c * 2);
  bf16* kr_b  = (bf16*)alloc((size_t)BHc * D2c * Tc * 2);
  float* ksum = (float*)alloc((size_t)BHc * D2c * 4);
  bf16* kvt_b = (bf16*)alloc((size_t)BHc * Dc * D2c * 2);
  float* zr   = (float*)alloc((size_t)BHc * Tc * 4);
  bf16* at_b  = (bf16*)alloc((size_t)BTc * Ec * 2);
  float* y1   = (float*)alloc((size_t)BTc * Ec * 4);
  bf16* h2_b  = (bf16*)alloc((size_t)BTc * Ec * 2);
  bf16* fc_b  = (bf16*)alloc((size_t)BTc * E4c * 2);

  const dim3 blk(256);
  typedef long long ll;

  // ---- weights fp32 -> bf16, transposed ----
  cvtT_kernel<<<dim3(Ec / 32, Ec / 32), blk, 0, stream>>>(wq, wq_t, Ec, Ec);
  cvtT_kernel<<<dim3(Ec / 32, Ec / 32), blk, 0, stream>>>(wk, wk_t, Ec, Ec);
  cvtT_kernel<<<dim3(Ec / 32, Ec / 32), blk, 0, stream>>>(wv, wv_t, Ec, Ec);
  cvtT_kernel<<<dim3(Ec / 32, Ec / 32), blk, 0, stream>>>(wo, wo_t, Ec, Ec);
  cvtT_kernel<<<dim3(E4c / 32, Ec / 32), blk, 0, stream>>>(wfc, wfc_t, Ec, E4c);
  cvtT_kernel<<<dim3(Ec / 32, E4c / 32), blk, 0, stream>>>(wpr, wpr_t, E4c, Ec);

  // ---- LN1 ----
  ln_kernel<<<dim3(BTc), blk, 0, stream>>>(x, ln1w, ln1b, h1_b);

  // ---- Q/K/V GEMMs ----
  const dim3 gE(Ec / BN, BTc / BM, 1);
  gemm_kernel<1><<<gE, blk, 0, stream>>>(h1_b, wq_t, bq, nullptr, nullptr,
      nullptr, q_b, BTc, Ec, Ec, Ec, Ec, Ec, Ec, 0, 1,
      0, 0, 0, 0, 0, 0, 0, 0, 0, 0);
  gemm_kernel<1><<<gE, blk, 0, stream>>>(h1_b, wk_t, bk, nullptr, nullptr,
      nullptr, k_b, BTc, Ec, Ec, Ec, Ec, Ec, Ec, 0, 1,
      0, 0, 0, 0, 0, 0, 0, 0, 0, 0);
  gemm_kernel<0><<<gE, blk, 0, stream>>>(h1_b, wv_t, bv, nullptr, nullptr,
      nullptr, v_b, BTc, Ec, Ec, Ec, Ec, Ec, Ec, 0, 1,
      0, 0, 0, 0, 0, 0, 0, 0, 0, 0);

  // ---- per-head V transpose, cosine reweighting, k column sums ----
  vtr_kernel<<<dim3(Tc / 32, Dc / 32, BHc), blk, 0, stream>>>(v_b, vt_b);
  reweight_kernel<<<dim3(BTc), blk, 0, stream>>>(q_b, k_b, qr_b, kr_b);
  colsum_kernel<<<dim3(BHc), dim3(128), 0, stream>>>(kr_b, ksum);

  // ---- kv[bh] = K'^T(128 x T) @ V(T x 64), stored transposed as kvt[64x128]
  gemm_kernel<0><<<dim3(1, 1, BHc), blk, 0, stream>>>(
      kr_b, vt_b, nullptr, nullptr, nullptr, nullptr, kvt_b,
      D2c, Dc, Tc, Tc, Tc, Dc, D2c, /*cbT=*/1, Hc,
      (ll)Hc * D2c * Tc, (ll)D2c * Tc,    // A (b,h)
      (ll)Hc * Dc * Tc, (ll)Dc * Tc,      // Bt = vt (b,h)
      0, 0,
      (ll)Hc * Dc * D2c, (ll)Dc * D2c,    // Cb = kvt (b,h)
      0, 0);

  // ---- denominator reciprocal ----
  zden_kernel<<<dim3(Tc / 256, BHc), blk, 0, stream>>>(qr_b, ksum, zr);

  // ---- numerator: attn[b,t,h*64+d] = (Q'[Tx128] @ kv[128x64]) * zr ----
  gemm_kernel<0><<<dim3(1, Tc / BM, BHc), blk, 0, stream>>>(
      qr_b, kvt_b, nullptr, nullptr, zr, nullptr, at_b,
      Tc, Dc, D2c, D2c, D2c, Ec, Ec, 0, Hc,
      (ll)Hc * Tc * D2c, (ll)Tc * D2c,    // A = qr
      (ll)Hc * Dc * D2c, (ll)Dc * D2c,    // Bt = kvt
      0, 0,
      (ll)Tc * Ec, (ll)Dc,                // Cb = attn slice of E
      (ll)Hc * Tc, (ll)Tc);               // rowscale = zr

  // ---- O projection + residual: y1 = attn@wo + bo + x ----
  gemm_kernel<0><<<gE, blk, 0, stream>>>(at_b, wo_t, bo, x, nullptr,
      y1, nullptr, BTc, Ec, Ec, Ec, Ec, Ec, Ec, 0, 1,
      0, 0, 0, 0, 0, 0, 0, 0, 0, 0);

  // ---- LN2 ----
  ln_kernel<<<dim3(BTc), blk, 0, stream>>>(y1, ln2w, ln2b, h2_b);

  // ---- FC + exact GELU ----
  gemm_kernel<2><<<dim3(E4c / BN, BTc / BM, 1), blk, 0, stream>>>(
      h2_b, wfc_t, bfc, nullptr, nullptr, nullptr, fc_b,
      BTc, E4c, Ec, Ec, Ec, E4c, E4c, 0, 1,
      0, 0, 0, 0, 0, 0, 0, 0, 0, 0);

  // ---- projection + bias + residual -> out ----
  gemm_kernel<0><<<gE, blk, 0, stream>>>(fc_b, wpr_t, bpr, y1, nullptr,
      out, nullptr, BTc, Ec, E4c, E4c, E4c, Ec, Ec, 0, 1,
      0, 0, 0, 0, 0, 0, 0, 0, 0, 0);
}
